// MambaBlock_61083024883755
// MI455X (gfx1250) — compile-verified, matrix operand order checked by reading, not run on hardware
//
#include <hip/hip_runtime.h>
#include <hip/hip_bf16.h>
#include <math.h>

typedef _Float16 f16;
typedef __attribute__((ext_vector_type(16))) _Float16 v16h;
typedef __attribute__((ext_vector_type(8)))  _Float16 v8h;
typedef __attribute__((ext_vector_type(8)))  float    v8f;

#define D_MODEL  256
#define D_STATE  16
#define D_CONVK  4
#define D_INNER  512
#define DT_RANK  16
#define BATCH    2
#define LSEQ     4096
#define MROWS    (BATCH * LSEQ)   // 8192 rows of the (B*L, C) activation matrix
#define NPROJ    64               // DT_RANK + 2*D_STATE = 48, padded to 64 for WMMA tiles

__device__ __forceinline__ float sigmoidf_(float v) { return 1.0f / (1.0f + __expf(-v)); }
__device__ __forceinline__ float siluf_(float v)    { return v * sigmoidf_(v); }
__device__ __forceinline__ float softplusf_(float v){ return (v > 20.0f) ? v : log1pf(__expf(v)); }

// ---------------------------------------------------------------------------
// LayerNorm over C=256 for each (b,l) row; x layout is (B, C, L) -> strided
// channel reads; writes f16 activations (M x 256) for the WMMA in_proj GEMM.
// ---------------------------------------------------------------------------
__global__ __launch_bounds__(256) void k_ln1(const float* __restrict__ x,
                                             const float* __restrict__ w,
                                             const float* __restrict__ b,
                                             f16* __restrict__ out) {
    int m = blockIdx.x;           // row in [0, MROWS)
    int c = threadIdx.x;          // channel
    int bb = m >> 12, l = m & (LSEQ - 1);
    float v = x[(size_t)bb * D_MODEL * LSEQ + (size_t)c * LSEQ + l];
    __shared__ float red[256];
    red[c] = v; __syncthreads();
    for (int s = 128; s > 0; s >>= 1) { if (c < s) red[c] += red[c + s]; __syncthreads(); }
    float mu = red[0] * (1.0f / 256.0f);
    __syncthreads();
    float d = v - mu;
    red[c] = d * d; __syncthreads();
    for (int s = 128; s > 0; s >>= 1) { if (c < s) red[c] += red[c + s]; __syncthreads(); }
    float var = red[0] * (1.0f / 256.0f);
    float r = rsqrtf(var + 1e-5f);
    out[(size_t)m * D_MODEL + c] = (f16)(d * r * w[c] + b[c]);
}

// f32 -> f16 weight conversion
__global__ __launch_bounds__(256) void k_cvt(const float* __restrict__ src,
                                             f16* __restrict__ dst, int n) {
    int i = blockIdx.x * 256 + threadIdx.x;
    if (i < n) dst[i] = (f16)src[i];
}

// f32 -> f16 with zero row padding (x_proj_w: 48x512 -> 64x512)
__global__ __launch_bounds__(256) void k_cvt_pad(const float* __restrict__ src,
                                                 f16* __restrict__ dst,
                                                 int rows, int cols, int padRows) {
    int i = blockIdx.x * 256 + threadIdx.x;
    int total = padRows * cols;
    if (i >= total) return;
    int r = i / cols;
    dst[i] = (r < rows) ? (f16)src[i] : (f16)0.0f;
}

// ---------------------------------------------------------------------------
// Wave-level 16x16 tile GEMM core: D += A(16xK) * W(16 rows of K)^T using
// v_wmma_f32_16x16x32_f16. Per-lane fragment loads match the CDNA5 16-bit
// A/B VGPR layout: lanes 0-15 hold K chunks [kb..kb+7],[kb+16..kb+23];
// lanes 16-31 hold [kb+8..kb+15],[kb+24..kb+31].
// ---------------------------------------------------------------------------
__device__ __forceinline__ v8f wmma_rowcol(const f16* __restrict__ A,
                                           const f16* __restrict__ W,
                                           int K, int aRow, int bRow, int half) {
    v8f acc = {0.f, 0.f, 0.f, 0.f, 0.f, 0.f, 0.f, 0.f};
    const f16* pa = A + (size_t)aRow * K + half * 8;
    const f16* pb = W + (size_t)bRow * K + half * 8;
    for (int kb = 0; kb < K; kb += 32) {
        union { v16h v; v8h h[2]; } ua, ub;
        ua.h[0] = *(const v8h*)(pa + kb);
        ua.h[1] = *(const v8h*)(pa + kb + 16);
        ub.h[0] = *(const v8h*)(pb + kb);
        ub.h[1] = *(const v8h*)(pb + kb + 16);
        if (kb + 32 < K) {
            __builtin_prefetch(pa + kb + 32, 0, 1);   // global_prefetch_b8
            __builtin_prefetch(pb + kb + 32, 0, 1);
        }
        acc = __builtin_amdgcn_wmma_f32_16x16x32_f16(false, ua.v, false, ub.v,
                                                     (short)0, acc, false, false);
    }
    return acc;
}

// in_proj: (8192x256) x (1024x256)^T -> split epilogue: cols<512 -> xm,
// cols>=512 -> silu(z) (gate pre-applied).
__global__ __launch_bounds__(256) void k_inproj(const f16* __restrict__ A,
                                                const f16* __restrict__ W,
                                                float* __restrict__ xm,
                                                float* __restrict__ zs) {
    const int K = D_MODEL, N = 2 * D_INNER;
    int wave = blockIdx.x * 8 + (threadIdx.x >> 5);
    int tilesN = N / 16;
    if (wave >= (MROWS / 16) * tilesN) return;
    int m0 = (wave / tilesN) * 16, n0 = (wave % tilesN) * 16;
    int lane = threadIdx.x & 31;
    int half = lane >> 4;
    v8f acc = wmma_rowcol(A, W, K, m0 + (lane & 15), n0 + (lane & 15), half);
    int col = n0 + (lane & 15);
    int mbase = m0 + half * 8;
    if (col < D_INNER) {
        #pragma unroll
        for (int v = 0; v < 8; ++v)
            xm[(size_t)(mbase + v) * D_INNER + col] = acc[v];
    } else {
        int cc = col - D_INNER;
        #pragma unroll
        for (int v = 0; v < 8; ++v)
            zs[(size_t)(mbase + v) * D_INNER + cc] = siluf_(acc[v]);
    }
}

// Generic WMMA GEMM, plain f32 store: C(MxN) = A(MxK_f16) * W(NxK_f16)^T
__global__ __launch_bounds__(256) void k_gemm(const f16* __restrict__ A,
                                              const f16* __restrict__ W,
                                              float* __restrict__ C,
                                              int M, int N, int K) {
    int wave = blockIdx.x * 8 + (threadIdx.x >> 5);
    int tilesN = N / 16;
    if (wave >= (M / 16) * tilesN) return;
    int m0 = (wave / tilesN) * 16, n0 = (wave % tilesN) * 16;
    int lane = threadIdx.x & 31;
    int half = lane >> 4;
    v8f acc = wmma_rowcol(A, W, K, m0 + (lane & 15), n0 + (lane & 15), half);
    int col = n0 + (lane & 15);
    int mbase = m0 + half * 8;
    #pragma unroll
    for (int v = 0; v < 8; ++v)
        C[(size_t)(mbase + v) * N + col] = acc[v];
}

// Depthwise causal conv1d (k=4) over L per channel + bias + SiLU.
__global__ __launch_bounds__(256) void k_conv(const float* __restrict__ xm,
                                              const float* __restrict__ cw,
                                              const float* __restrict__ cb,
                                              float* __restrict__ xc,
                                              f16* __restrict__ xc16) {
    int i = blockIdx.x * 256 + threadIdx.x;
    if (i >= MROWS * D_INNER) return;
    int m = i / D_INNER, d = i - m * D_INNER;
    int l = m & (LSEQ - 1);
    float acc = cb[d];
    #pragma unroll
    for (int j = 0; j < D_CONVK; ++j) {
        int lj = l + j - (D_CONVK - 1);
        if (lj >= 0)
            acc += cw[d * D_CONVK + j] * xm[(size_t)(m + j - (D_CONVK - 1)) * D_INNER + d];
    }
    float s = siluf_(acc);
    xc[i]   = s;
    xc16[i] = (f16)s;
}

// dt = softplus(proj[:, :16] @ dt_proj_w^T + dt_bias); K=16 -> plain VALU.
__global__ __launch_bounds__(256) void k_dt(const float* __restrict__ proj,
                                            const float* __restrict__ wdt,
                                            const float* __restrict__ bdt,
                                            float* __restrict__ dtv) {
    int i = blockIdx.x * 256 + threadIdx.x;
    if (i >= MROWS * D_INNER) return;
    int m = i / D_INNER, d = i - m * D_INNER;
    float acc = bdt[d];
    #pragma unroll
    for (int r = 0; r < DT_RANK; ++r)
        acc += proj[(size_t)m * NPROJ + r] * wdt[d * DT_RANK + r];
    dtv[i] = softplusf_(acc);
}

// Selective scan: one thread per (b,d); 16-state recurrence in registers.
// Fuses skip (x*D) and gate (silu(z), precomputed) and emits f16 for out_proj.
__global__ __launch_bounds__(512) void k_scan(const float* __restrict__ xc,
                                              const float* __restrict__ dtv,
                                              const float* __restrict__ proj,
                                              const float* __restrict__ A_log,
                                              const float* __restrict__ Dp,
                                              const float* __restrict__ zs,
                                              f16* __restrict__ y16) {
    int b = blockIdx.x;
    int d = threadIdx.x;
    float Aa[D_STATE];
    #pragma unroll
    for (int s = 0; s < D_STATE; ++s) Aa[s] = -__expf(A_log[d * D_STATE + s]);
    float Dd = Dp[d];
    float h[D_STATE];
    #pragma unroll
    for (int s = 0; s < D_STATE; ++s) h[s] = 0.0f;
    for (int l = 0; l < LSEQ; ++l) {
        int m = b * LSEQ + l;
        float xt  = xc[(size_t)m * D_INNER + d];
        float dtt = dtv[(size_t)m * D_INNER + d];
        float dtx = dtt * xt;
        const float* pr = proj + (size_t)m * NPROJ;
        float acc = 0.0f;
        #pragma unroll
        for (int s = 0; s < D_STATE; ++s) {
            float dA = __expf(dtt * Aa[s]);
            h[s] = h[s] * dA + dtx * pr[DT_RANK + s];
            acc += h[s] * pr[DT_RANK + D_STATE + s];
        }
        float yv = (acc + xt * Dd) * zs[(size_t)m * D_INNER + d];
        y16[(size_t)m * D_INNER + d] = (f16)yv;
    }
}

// LN2 over 256 + residual; scatter back to (B, C, H, W).
__global__ __launch_bounds__(256) void k_ln2(const float* __restrict__ mamba,
                                             const float* __restrict__ w,
                                             const float* __restrict__ b,
                                             const float* __restrict__ x,
                                             float* __restrict__ out) {
    int m = blockIdx.x;
    int c = threadIdx.x;
    float v = mamba[(size_t)m * D_MODEL + c];
    __shared__ float red[256];
    red[c] = v; __syncthreads();
    for (int s = 128; s > 0; s >>= 1) { if (c < s) red[c] += red[c + s]; __syncthreads(); }
    float mu = red[0] * (1.0f / 256.0f);
    __syncthreads();
    float dd = v - mu;
    red[c] = dd * dd; __syncthreads();
    for (int s = 128; s > 0; s >>= 1) { if (c < s) red[c] += red[c + s]; __syncthreads(); }
    float var = red[0] * (1.0f / 256.0f);
    float r = rsqrtf(var + 1e-5f);
    float nrm = dd * r * w[c] + b[c];
    int bb = m >> 12, l = m & (LSEQ - 1);
    size_t oi = (size_t)bb * D_MODEL * LSEQ + (size_t)c * LSEQ + l;
    out[oi] = x[oi] + nrm;
}

extern "C" void kernel_launch(void* const* d_in, const int* in_sizes, int n_in,
                              void* d_out, int out_size, void* d_ws, size_t ws_size,
                              hipStream_t stream) {
    const float* x        = (const float*)d_in[0];
    const float* ln1_w    = (const float*)d_in[1];
    const float* ln1_b    = (const float*)d_in[2];
    const float* ln2_w    = (const float*)d_in[3];
    const float* ln2_b    = (const float*)d_in[4];
    const float* in_proj  = (const float*)d_in[5];   // (1024, 256)
    const float* conv_w   = (const float*)d_in[6];   // (512, 4)
    const float* conv_b   = (const float*)d_in[7];
    const float* x_proj   = (const float*)d_in[8];   // (48, 512)
    const float* dt_w     = (const float*)d_in[9];   // (512, 16)
    const float* dt_b     = (const float*)d_in[10];
    const float* A_log    = (const float*)d_in[11];  // (512, 16)
    const float* Dp       = (const float*)d_in[12];
    const float* out_proj = (const float*)d_in[13];  // (256, 512)
    float* out = (float*)d_out;

    char* ws = (char*)d_ws;
    size_t off = 0;
    auto take = [&](size_t bytes) { char* p = ws + off; off = (off + bytes + 255) & ~(size_t)255; return p; };

    f16*   hln16   = (f16*)  take((size_t)MROWS * D_MODEL * 2);      // LN1 output, f16
    f16*   w_in16  = (f16*)  take((size_t)2 * D_INNER * D_MODEL * 2);
    f16*   w_xp16  = (f16*)  take((size_t)NPROJ * D_INNER * 2);      // padded 64x512
    f16*   w_out16 = (f16*)  take((size_t)D_MODEL * D_INNER * 2);
    float* xm      = (float*)take((size_t)MROWS * D_INNER * 4);
    float* zs      = (float*)take((size_t)MROWS * D_INNER * 4);      // silu(z)
    float* xc      = (float*)take((size_t)MROWS * D_INNER * 4);      // conv+silu, f32
    f16*   xc16    = (f16*)  take((size_t)MROWS * D_INNER * 2);
    float* proj    = (float*)take((size_t)MROWS * NPROJ * 4);        // [dt | B | C | pad]
    float* dtv     = (float*)take((size_t)MROWS * D_INNER * 4);
    f16*   y16     = (f16*)  take((size_t)MROWS * D_INNER * 2);
    float* mamba   = (float*)take((size_t)MROWS * D_MODEL * 4);
    (void)ws_size; (void)in_sizes; (void)n_in; (void)out_size;

    // Weight conversions (f32 -> f16)
    k_cvt    <<<(2 * D_INNER * D_MODEL + 255) / 256, 256, 0, stream>>>(in_proj,  w_in16,  2 * D_INNER * D_MODEL);
    k_cvt_pad<<<(NPROJ * D_INNER + 255) / 256,       256, 0, stream>>>(x_proj,   w_xp16,  DT_RANK + 2 * D_STATE, D_INNER, NPROJ);
    k_cvt    <<<(D_MODEL * D_INNER + 255) / 256,     256, 0, stream>>>(out_proj, w_out16, D_MODEL * D_INNER);

    // 1) LayerNorm1 -> f16 activations
    k_ln1<<<MROWS, 256, 0, stream>>>(x, ln1_w, ln1_b, hln16);

    // 2) in_proj GEMM (WMMA) with split xm / silu(z) epilogue
    {
        int tiles = (MROWS / 16) * ((2 * D_INNER) / 16);     // 32768
        k_inproj<<<tiles / 8, 256, 0, stream>>>(hln16, w_in16, xm, zs);
    }

    // 3) depthwise causal conv + SiLU
    k_conv<<<(MROWS * D_INNER + 255) / 256, 256, 0, stream>>>(xm, conv_w, conv_b, xc, xc16);

    // 4) x_proj GEMM (WMMA), N padded to 64
    {
        int tiles = (MROWS / 16) * (NPROJ / 16);             // 2048
        k_gemm<<<tiles / 8, 256, 0, stream>>>(xc16, w_xp16, proj, MROWS, NPROJ, D_INNER);
    }

    // 5) dt projection + softplus
    k_dt<<<(MROWS * D_INNER + 255) / 256, 256, 0, stream>>>(proj, dt_w, dt_b, dtv);

    // 6) selective scan + skip + gate, emit f16
    k_scan<<<BATCH, D_INNER, 0, stream>>>(xc, dtv, proj, A_log, Dp, zs, y16);

    // 7) out_proj GEMM (WMMA)
    {
        int tiles = (MROWS / 16) * (D_MODEL / 16);           // 8192
        k_gemm<<<tiles / 8, 256, 0, stream>>>(y16, w_out16, mamba, MROWS, D_MODEL, D_INNER);
    }

    // 8) LayerNorm2 + residual, scatter to (B, C, H, W)
    k_ln2<<<MROWS, 256, 0, stream>>>(mamba, ln2_w, ln2_b, x, out);
}
